// MultiHeadAttention_34531537060226
// MI455X (gfx1250) — matrix operand order repaired and checked by reading, not compiled, not run on hardware
//
#include <hip/hip_runtime.h>

// ---------------------------------------------------------------------------
// Types
// ---------------------------------------------------------------------------
typedef __bf16 bf16;
typedef __attribute__((ext_vector_type(16))) __bf16 v16bf;
typedef __attribute__((ext_vector_type(8)))  __bf16 bf16x8;
typedef __attribute__((ext_vector_type(4)))  __bf16 bf16x4;
typedef __attribute__((ext_vector_type(2)))  __bf16 bf16x2;
typedef __attribute__((ext_vector_type(8)))  float  v8f;
typedef __attribute__((ext_vector_type(4)))  float  f32x4;
typedef __attribute__((ext_vector_type(4)))  int    i32x4;

// Problem constants (from the reference)
constexpr int T_SEQ   = 2048;
constexpr int D_MODEL = 1024;
constexpr int NHEADS  = 16;
constexpr int DHEAD   = 64;
constexpr int BATCH   = 2;
constexpr int M_ROWS  = BATCH * T_SEQ;   // 4096 tokens

#define WMMA_BF16(a, b, c)                                                    \
  __builtin_amdgcn_wmma_f32_16x16x32_bf16(false, (a), false, (b), (short)0,   \
                                          (c), false, false)

// ---------------------------------------------------------------------------
// Optional gfx1250 async global->LDS staging (ASYNCcnt path).
// Guarded by __has_builtin so the file compiles on toolchains without it.
// Probe result (round 3 diagnostic): builtin exists; param 0 is
// addrspace(1) int4* ("__device__" in HIP diag printing), pointee must be int4.
// ---------------------------------------------------------------------------
#if defined(__has_builtin)
#if __has_builtin(__builtin_amdgcn_global_load_async_to_lds_b128)
#define HAVE_ASYNC_LDS 1
#endif
#if __has_builtin(__builtin_amdgcn_s_wait_asynccnt)
#define HAVE_WAIT_ASYNC 1
#endif
#endif

typedef __attribute__((address_space(1))) i32x4 g1_i32x4;
typedef __attribute__((address_space(3))) i32x4 l3_i32x4;

#if defined(HAVE_ASYNC_LDS)
__device__ __forceinline__ void async_copy_b128(const void* gsrc, void* ldst) {
  // generic->local addrspacecast == low 32 bits of the flat address.
  __builtin_amdgcn_global_load_async_to_lds_b128(
      (g1_i32x4*)(unsigned long long)gsrc,
      (l3_i32x4*)(unsigned)(unsigned long long)ldst, 0, 0);
}
__device__ __forceinline__ void async_wait0() {
#if defined(HAVE_WAIT_ASYNC)
  __builtin_amdgcn_s_wait_asynccnt(0);
#else
  asm volatile("s_wait_asynccnt 0x0" ::: "memory");
#endif
}
#endif

// ---------------------------------------------------------------------------
// Fragment assembly per CDNA5 ISA 7.12.2 (wave32):
//   A (16x32 bf16): lane m = lane&15; VGPR v holds K pair at
//                   k0 = (v/4)*16 + (v%4)*2 + (lane/16)*8
//   B (32x16 bf16): lane n = lane&15; VGPR v holds K pair at
//                   k0 = (v/4)*16 + (v%4)*2 + (lane/16)*8 (source stored n-major)
//   C/D (16x16 f32): VGPR v -> row m = v + 8*(lane/16); col n = lane&15
// (The compiler merges each fragment into 2x ds_load_b128.)
// ---------------------------------------------------------------------------
__device__ __forceinline__ v16bf load_frag_a(const bf16* p, int stride, int lane) {
  const int m = lane & 15, half = lane >> 4;
  v16bf a;
#pragma unroll
  for (int v = 0; v < 8; ++v) {
    const int k0 = ((v >> 2) << 4) + ((v & 3) << 1) + (half << 3);
    bf16x2 pr = *(const bf16x2*)(p + m * stride + k0);
    a[2 * v] = pr.x;
    a[2 * v + 1] = pr.y;
  }
  return a;
}

__device__ __forceinline__ v16bf load_frag_b(const bf16* p, int stride, int lane) {
  const int n = lane & 15, half = lane >> 4;
  v16bf b;
#pragma unroll
  for (int v = 0; v < 8; ++v) {
    const int k0 = ((v >> 2) << 4) + ((v & 3) << 1) + (half << 3);
    bf16x2 pr = *(const bf16x2*)(p + n * stride + k0);
    b[2 * v] = pr.x;
    b[2 * v + 1] = pr.y;
  }
  return b;
}

// ---------------------------------------------------------------------------
// 16-lane max reduction entirely in VALU via DPP16 (no ds_bpermute).
// ---------------------------------------------------------------------------
template <int CTRL>
__device__ __forceinline__ float dpp_fmax(float x) {
  int xi = __builtin_bit_cast(int, x);
  int yi = __builtin_amdgcn_update_dpp(xi, xi, CTRL, 0xF, 0xF, true);
  return fmaxf(x, __builtin_bit_cast(float, yi));
}

__device__ __forceinline__ float row16_max(float x) {
  x = dpp_fmax<0xB1>(x);   // quad_perm(1,0,3,2): swap within pairs
  x = dpp_fmax<0x4E>(x);   // quad_perm(2,3,0,1): quad uniform
  x = dpp_fmax<0x141>(x);  // row_half_mirror: octet uniform
  x = dpp_fmax<0x140>(x);  // row_mirror: 16 lanes uniform
  return x;
}

// ---------------------------------------------------------------------------
// GEMM + bias:  Y[m,n] = (sum_k X[m,k] * W[k,n] + bias[n]) * alpha
//   X: fp32 or bf16, row-major [M_ROWS, D_MODEL]
//   W: fp32 row-major [D_MODEL, D_MODEL]  (converted to bf16 in LDS)
//   STORE == 0: write bf16 head-split [B, H, T, Dh] (for Q/K/V projections)
//   STORE == 1: write fp32 flat [M_ROWS, D_MODEL]   (final output proj)
// Block: 256 threads = 8 waves; tile BM=64, BN=128, BK=32.
// Each wave: 1 A-frag + 4 B-frags + 4 WMMAs per K-step.
// bf16 X (output projection) stages its A tile with async global->LDS copies.
// ---------------------------------------------------------------------------
template <typename XT, int STORE>
__global__ __launch_bounds__(256) void gemm_bias_kernel(
    const XT* __restrict__ X, const float* __restrict__ W,
    const float* __restrict__ bias, void* __restrict__ outp, float alpha) {
  constexpr int BM = 64, BN = 128, BK = 32;
  // SA: byte stride 80 (16B aligned rows -> async b128 dst + ds_load_b128 ok)
  constexpr int SA = 40, SB = 36;
  __shared__ bf16 As[BM * SA];     // m-major
  __shared__ bf16 Bs[BN * SB];     // n-major (transposed on fill)

  const int tid  = threadIdx.x;
  const int lane = tid & 31;
  const int wid  = tid >> 5;        // 0..7
  const int wr   = wid >> 1;        // 0..3 -> C rows 16*wr
  const int wc   = wid & 1;         // 0..1 -> C cols 64*wc
  const int half = lane >> 4;
  const int nl   = lane & 15;
  const int m0   = blockIdx.y * BM;
  const int n0   = blockIdx.x * BN;

  v8f acc[4] = {v8f{}, v8f{}, v8f{}, v8f{}};

  const int ar = tid >> 2;          // 0..63 : A tile row
  const int ac = (tid & 3) * 8;     // A tile col start (8 elems = 16B bf16)
  const int br = tid >> 3;          // 0..31 : B tile k-row
  const int bc = (tid & 7) * 16;    // B tile n start (16 elems)

  for (int kk = 0; kk < D_MODEL; kk += BK) {
    // ---- stage A tile ----
    {
      const XT* src = X + (size_t)(m0 + ar) * D_MODEL + kk + ac;
      bool staged = false;
#if defined(HAVE_ASYNC_LDS)
      if constexpr (sizeof(XT) == 2) {
        async_copy_b128((const void*)src, (void*)&As[ar * SA + ac]);
        staged = true;
      }
#endif
      if (!staged) {
        bf16 tmp[8];
        if constexpr (sizeof(XT) == 4) {
          f32x4 x0 = *(const f32x4*)(src);
          f32x4 x1 = *(const f32x4*)(src + 4);
#pragma unroll
          for (int j = 0; j < 4; ++j) {
            tmp[j]     = (bf16)x0[j];
            tmp[4 + j] = (bf16)x1[j];
          }
        } else {
          bf16x8 x = *(const bf16x8*)(src);
#pragma unroll
          for (int j = 0; j < 8; ++j) tmp[j] = x[j];
        }
        bf16x4 lo = {tmp[0], tmp[1], tmp[2], tmp[3]};
        bf16x4 hi = {tmp[4], tmp[5], tmp[6], tmp[7]};
        *(bf16x4*)(&As[ar * SA + ac])     = lo;
        *(bf16x4*)(&As[ar * SA + ac + 4]) = hi;
      }
      if (kk + BK < D_MODEL)  // gfx1250 global_prefetch_b8 on next A tile
        __builtin_prefetch((const void*)(src + BK), 0, 1);
    }
    // ---- stage B tile transposed (n-major), convert to bf16 ----
    {
      const float* src = W + (size_t)(kk + br) * D_MODEL + n0 + bc;
#pragma unroll
      for (int q = 0; q < 4; ++q) {
        f32x4 x = *(const f32x4*)(src + 4 * q);
#pragma unroll
        for (int j = 0; j < 4; ++j)
          Bs[(bc + 4 * q + j) * SB + br] = (bf16)x[j];
      }
    }
#if defined(HAVE_ASYNC_LDS)
    if constexpr (sizeof(XT) == 2) async_wait0();
#endif
    __syncthreads();

    v16bf a = load_frag_a(&As[(wr * 16) * SA], SA, lane);
#pragma unroll
    for (int s = 0; s < 4; ++s) {
      v16bf b = load_frag_b(&Bs[(wc * 64 + s * 16) * SB], SB, lane);
      acc[s] = WMMA_BF16(a, b, acc[s]);
    }

    __syncthreads();
  }

  // ---- epilogue: bias add, alpha scale, store ----
#pragma unroll
  for (int s = 0; s < 4; ++s) {
    const int   gn = n0 + wc * 64 + s * 16 + nl;
    const float bv = bias[gn];
#pragma unroll
    for (int v = 0; v < 8; ++v) {
      const int   gm  = m0 + wr * 16 + v + half * 8;
      const float val = (acc[s][v] + bv) * alpha;
      if constexpr (STORE == 0) {
        const int b = gm / T_SEQ, t = gm % T_SEQ;
        const int h = gn / DHEAD, dh = gn % DHEAD;
        ((bf16*)outp)[(((size_t)b * NHEADS + h) * T_SEQ + t) * DHEAD + dh] =
            (bf16)val;
      } else {
        ((float*)outp)[(size_t)gm * D_MODEL + gn] = val;
      }
    }
  }
}

// ---------------------------------------------------------------------------
// Flash attention with online softmax.
//   Qh (pre-scaled by 1/sqrt(Dh)) / Kh / Vh: bf16 [B*H, T, Dh]
//   Ctx: bf16 [B*T, D_MODEL] (token-major)
// Block: 256 threads = 8 waves; each wave owns 16 query rows (128/block).
// Per 64-key chunk: 8 WMMAs scores, DPP row-max, exp, P->LDS,
// 2 WMMAs row-sum (P @ ones), 8 WMMAs PV.
// K chunk is staged with async global->LDS b128 copies when available
// (row byte-stride 144 = 9*16 keeps every b128 destination 16B aligned).
// ---------------------------------------------------------------------------
__global__ __launch_bounds__(256) void attn_kernel(
    const bf16* __restrict__ Qh, const bf16* __restrict__ Kh,
    const bf16* __restrict__ Vh, bf16* __restrict__ Ctx) {
  constexpr int KC = 64;  // keys per chunk
  constexpr int SK = 72;  // ldsK[key][dh]   (n-major, n=key)
  constexpr int SV = 66;  // ldsVt[dh][key]  (n-major, n=dh)
  constexpr int SP = 66;  // ldsP[m][k]      (m-major, k=key)
  __shared__ bf16 ldsK[KC * SK];
  __shared__ bf16 ldsVt[DHEAD * SV];
  __shared__ bf16 ldsP[8 * 16 * SP];

  const int tid  = threadIdx.x;
  const int lane = tid & 31;
  const int wid  = tid >> 5;
  const int half = lane >> 4;
  const int nl   = lane & 15;

  const int    bh   = blockIdx.y;                   // batch*head
  const size_t base = (size_t)bh * T_SEQ * DHEAD;
  const int    q0   = blockIdx.x * 128 + wid * 16;  // wave's first query row

  // Resident Q fragments: rows q0..q0+15, split dh 0..31 / 32..63
  v16bf qA0, qA1;
  {
    const bf16* Qg = Qh + base + (size_t)(q0 + nl) * DHEAD;
#pragma unroll
    for (int v = 0; v < 8; ++v) {
      const int k0 = ((v >> 2) << 4) + ((v & 3) << 1) + (half << 3);
      bf16x2 p0 = *(const bf16x2*)(Qg + k0);
      bf16x2 p1 = *(const bf16x2*)(Qg + 32 + k0);
      qA0[2 * v] = p0.x; qA0[2 * v + 1] = p0.y;
      qA1[2 * v] = p1.x; qA1[2 * v + 1] = p1.y;
    }
  }

  // Constant all-ones B fragment: row-sum of P via WMMA.
  v16bf onesB;
#pragma unroll
  for (int i = 0; i < 16; ++i) onesB[i] = (bf16)1.0f;

  float mrun[8], lrun[8];
#pragma unroll
  for (int v = 0; v < 8; ++v) { mrun[v] = -3.0e38f; lrun[v] = 0.0f; }
  v8f acc[4] = {v8f{}, v8f{}, v8f{}, v8f{}};

  const int kr = tid >> 2;        // 0..63: key row within chunk
  const int lc = (tid & 3) * 16;  // dh start (16 elems)

  for (int kc = 0; kc < T_SEQ; kc += KC) {
    // ---- stage K chunk (key-major == n-major for score B frags) ----
    {
      const bf16* src = Kh + base + (size_t)(kc + kr) * DHEAD + lc;
      bf16* dst = &ldsK[kr * SK + lc];
#if defined(HAVE_ASYNC_LDS)
      async_copy_b128((const void*)src, (void*)dst);
      async_copy_b128((const void*)(src + 8), (void*)(dst + 8));
#else
      bf16x8 x0 = *(const bf16x8*)(src);
      bf16x8 x1 = *(const bf16x8*)(src + 8);
      bf16x4 a0 = {x0[0], x0[1], x0[2], x0[3]};
      bf16x4 a1 = {x0[4], x0[5], x0[6], x0[7]};
      bf16x4 a2 = {x1[0], x1[1], x1[2], x1[3]};
      bf16x4 a3 = {x1[4], x1[5], x1[6], x1[7]};
      *(bf16x4*)(dst)      = a0;
      *(bf16x4*)(dst + 4)  = a1;
      *(bf16x4*)(dst + 8)  = a2;
      *(bf16x4*)(dst + 12) = a3;
#endif
    }
    // ---- stage V chunk transposed (dh-major for PV B frags) ----
    {
      const bf16* src = Vh + base + (size_t)(kc + kr) * DHEAD + lc;
      bf16x8 x0 = *(const bf16x8*)(src);
      bf16x8 x1 = *(const bf16x8*)(src + 8);
#pragma unroll
      for (int j = 0; j < 8; ++j) {
        ldsVt[(lc + j) * SV + kr]     = x0[j];
        ldsVt[(lc + 8 + j) * SV + kr] = x1[j];
      }
    }
#if defined(HAVE_ASYNC_LDS)
    async_wait0();
#endif
    __syncthreads();

    // ---- scores: 16 queries x 64 keys, Dh=64 contraction ----
    v8f c[4] = {v8f{}, v8f{}, v8f{}, v8f{}};
#pragma unroll
    for (int s = 0; s < 4; ++s) {  // key subtiles of 16
      v16bf blo = load_frag_b(&ldsK[(s * 16) * SK], SK, lane);       // dh 0-31
      c[s] = WMMA_BF16(qA0, blo, c[s]);
      v16bf bhi = load_frag_b(&ldsK[(s * 16) * SK + 32], SK, lane);  // dh 32-63
      c[s] = WMMA_BF16(qA1, bhi, c[s]);
    }

    // ---- online softmax: DPP row-max, exp, P -> LDS ----
    float corr[8];
#pragma unroll
    for (int v = 0; v < 8; ++v) {
      const float s0 = c[0][v], s1 = c[1][v], s2 = c[2][v], s3 = c[3][v];
      float r = fmaxf(fmaxf(s0, s1), fmaxf(s2, s3));
      r = row16_max(r);
      const float mnew = fmaxf(mrun[v], r);
      corr[v] = __expf(mrun[v] - mnew);
      mrun[v] = mnew;
      const float p0 = __expf(s0 - mnew);
      const float p1 = __expf(s1 - mnew);
      const float p2 = __expf(s2 - mnew);
      const float p3 = __expf(s3 - mnew);
      bf16* pr = &ldsP[((wid * 16) + v + half * 8) * SP];
      pr[nl]      = (bf16)p0;
      pr[nl + 16] = (bf16)p1;
      pr[nl + 32] = (bf16)p2;
      pr[nl + 48] = (bf16)p3;
#pragma unroll
      for (int g = 0; g < 4; ++g) acc[g][v] *= corr[v];
    }

    // ---- P fragments (16x64 = two A frags) ----
    v16bf pA0 = load_frag_a(&ldsP[wid * 16 * SP], SP, lane);       // keys 0-31
    v16bf pA1 = load_frag_a(&ldsP[wid * 16 * SP + 32], SP, lane);  // keys 32-63

    // ---- row-sum of P via WMMA against ones ----
    v8f rs = {};
    rs = WMMA_BF16(pA0, onesB, rs);
    rs = WMMA_BF16(pA1, onesB, rs);
#pragma unroll
    for (int v = 0; v < 8; ++v) lrun[v] = lrun[v] * corr[v] + rs[v];

    // ---- PV: ctx += P(16x64) @ V(64x64) ----
#pragma unroll
    for (int g = 0; g < 4; ++g) {  // dh groups of 16
      v16bf vb0 = load_frag_b(&ldsVt[(g * 16) * SV], SV, lane);       // keys 0-31
      acc[g] = WMMA_BF16(pA0, vb0, acc[g]);
      v16bf vb1 = load_frag_b(&ldsVt[(g * 16) * SV + 32], SV, lane);  // keys 32-63
      acc[g] = WMMA_BF16(pA1, vb1, acc[g]);
    }
    __syncthreads();
  }

  // ---- epilogue: normalize and scatter to token-major ctx (bf16) ----
  const int b = bh / NHEADS, h = bh % NHEADS;
#pragma unroll
  for (int v = 0; v < 8; ++v) {
    const int   t   = q0 + v + half * 8;
    const float inv = 1.0f / lrun[v];
#pragma unroll
    for (int g = 0; g < 4; ++g) {
      Ctx[(size_t)(b * T_SEQ + t) * D_MODEL + h * DHEAD + g * 16 + nl] =
          (bf16)(acc[g][v] * inv);
    }
  }
}

// ---------------------------------------------------------------------------
// Launch
//   d_in: q,k,v, Wq,bq, Wk,bk, Wv,bv, Wo,bo  (all fp32)
//   d_out: fp32 [B, T, D_MODEL]
//   d_ws:  bf16 Qh | Kh | Vh (head-split) | Ctx (token-major)  = ~33.6 MB
// ---------------------------------------------------------------------------
extern "C" void kernel_launch(void* const* d_in, const int* in_sizes, int n_in,
                              void* d_out, int out_size, void* d_ws,
                              size_t ws_size, hipStream_t stream) {
  const float* q  = (const float*)d_in[0];
  const float* k  = (const float*)d_in[1];
  const float* v  = (const float*)d_in[2];
  const float* Wq = (const float*)d_in[3];
  const float* bq = (const float*)d_in[4];
  const float* Wk = (const float*)d_in[5];
  const float* bk = (const float*)d_in[6];
  const float* Wv = (const float*)d_in[7];
  const float* bv = (const float*)d_in[8];
  const float* Wo = (const float*)d_in[9];
  const float* bo = (const float*)d_in[10];
  float* out = (float*)d_out;

  const size_t planeElems = (size_t)M_ROWS * D_MODEL;  // 4M elems
  bf16* Qh  = (bf16*)d_ws;
  bf16* Kh  = Qh + planeElems;
  bf16* Vh  = Kh + planeElems;
  bf16* Ctx = Vh + planeElems;

  const dim3 gGemm(D_MODEL / 128, M_ROWS / 64);  // (8, 64)
  const dim3 bGemm(256);

  // Q/K/V projections: fp32 in, head-split bf16 out.
  // Q is pre-scaled by 1/sqrt(Dh) so attention never multiplies scores.
  gemm_bias_kernel<float, 0><<<gGemm, bGemm, 0, stream>>>(q, Wq, bq, (void*)Qh,
                                                          0.125f);
  gemm_bias_kernel<float, 0><<<gGemm, bGemm, 0, stream>>>(k, Wk, bk, (void*)Kh,
                                                          1.0f);
  gemm_bias_kernel<float, 0><<<gGemm, bGemm, 0, stream>>>(v, Wv, bv, (void*)Vh,
                                                          1.0f);

  // Attention: 16 query-blocks x 32 (batch*head)
  attn_kernel<<<dim3(T_SEQ / 128, BATCH * NHEADS), dim3(256), 0, stream>>>(
      Qh, Kh, Vh, Ctx);

  // Output projection: bf16 ctx in, fp32 out + bias
  gemm_bias_kernel<bf16, 1><<<gGemm, bGemm, 0, stream>>>(Ctx, Wo, bo,
                                                         (void*)out, 1.0f);
}
